// GraphAttentionLayer_30090540876453
// MI455X (gfx1250) — matrix operand order repaired and checked by reading, not compiled, not run on hardware
//
#include <hip/hip_runtime.h>

// GAT layer as masked flash-attention on gfx1250 (wave32, WMMA + TDM).
// B=2, S=4096, F=128, H=4, head-dim Kd=32, LeakyReLU alpha=0.2.
// Pipeline: pre-convert x to f16 in workspace; per chunk each wave DMAs its
// 16x32 f16 tile into LDS via TENSOR_LOAD_TO_LDS with hardware row padding,
// double-buffered and synchronized with s_wait_tensorcnt.

typedef __attribute__((ext_vector_type(16))) _Float16 v16h;
typedef __attribute__((ext_vector_type(4)))  _Float16 v4h;
typedef __attribute__((ext_vector_type(8)))  float    v8f;
typedef __attribute__((ext_vector_type(4)))  float    v4f;
typedef __attribute__((ext_vector_type(4)))  unsigned int v4u;
typedef __attribute__((ext_vector_type(4)))  int      v4i;
typedef __attribute__((ext_vector_type(8)))  int      v8i;

#define SEQ   4096
#define FDIM  128
#define NH    4
#define KD    32
#define JT    64            // keys per chunk
#define NCH   (SEQ / JT)
#define LOG2E 1.4426950408889634f
#define NEGBIG (-3.0e38f)
// leaky_relu(s) = 0.6s + 0.4|s| ; log2e folded in for the TDM kernel
#define C1  0.6f
#define C2  0.4f
#define C1L (0.6f * LOG2E)
#define C2L (0.4f * LOG2E)

#define SXS 36              // sx row stride (halves): 72B = 64B row + 8B pad
#define SPS 70              // sp row stride (halves): disjoint bank sets

// ---- WMMA fragment index maps (contiguous k = e + 16*hi, same for A and B, wave32) ----
__device__ __forceinline__ int amapK(int hi, int e) {
  return e + (hi ? 16 : 0);
}
__device__ __forceinline__ int bmapK(int hi, int e) {
  return e + (hi ? 16 : 0);
}

__device__ __forceinline__ float rowmax16(float v) {
#pragma unroll
  for (int off = 1; off < 16; off <<= 1)
    v = fmaxf(v, __shfl_xor(v, off, 16));
  return v;
}
__device__ __forceinline__ float rowsum16(float v) {
#pragma unroll
  for (int off = 1; off < 16; off <<= 1)
    v += __shfl_xor(v, off, 16);
  return v;
}

// ---- TDM: DMA a 16-row x 32-half tile into LDS with 8B row padding ----
// D# per CDNA5 ISA (2D, data_size=2B,
// tensor_dim0=32, tensor_dim0_stride=128, tile 32x16, pad 16dw->2dw).
__device__ __forceinline__ void tdm_load_16x32(unsigned lds_byte,
                                               const _Float16* gptr) {
  unsigned long long ga = (unsigned long long)(size_t)gptr;
  v4u g0;
  g0[0] = 1u;                                            // count=1, user D#
  g0[1] = lds_byte;                                      // lds_addr (bytes)
  g0[2] = (unsigned)ga;                                  // global_addr[31:0]
  g0[3] = (unsigned)((ga >> 32) & 0x01ffffffu) | (2u << 30);  // [56:32]|type=2
  v8i g1;
  g1[0] = (int)((1u << 16)        // data_size = 1 -> 2 bytes
              | (1u << 20)        // pad_enable
              | (3u << 22)        // pad_interval = 3 -> every 16 DWORDs (64B)
              | (1u << 25));      // pad_amount  = 1 -> 2 DWORDs (8B)
  g1[1] = (int)(32u << 16);       // tensor_dim0[15:0] = 32 (bits 63:48)
  g1[2] = (int)((unsigned)SEQ << 16);  // tensor_dim1[15:0] (bits 95:80)
  g1[3] = (int)(32u << 16);       // tile_dim0 = 32 (bits 127:112)
  g1[4] = 16;                     // tile_dim1 = 16 rows; tile_dim2 = 0
  g1[5] = FDIM;                   // tensor_dim0_stride[31:0] = 128 elems
  g1[6] = 0;
  g1[7] = 0;
  v4i z4 = {};
  v8i z8 = {};
  __builtin_amdgcn_tensor_load_to_lds(g0, g1, z4, z4, z8, 0);
}

// ------------------- TDM main kernel (f16 x from workspace) ---------------
__global__ __launch_bounds__(128)
void gat_flash_wmma_tdm(const _Float16* __restrict__ xh,
                        const int*      __restrict__ adj,
                        float*          __restrict__ out) {
  __shared__ _Float16 sx[2][JT][SXS];    // double-buffered K/V tile
  __shared__ _Float16 sp[4][16][SPS];    // per-wave P tile

  const int lane = threadIdx.x & 31;
  const int w    = threadIdx.x >> 5;
  const int hi   = lane >> 4;
  const int ln   = lane & 15;

  const int qb = blockIdx.x & 63;
  const int h  = (blockIdx.x >> 6) & (NH - 1);
  const int b  = blockIdx.x >> 8;

  const int i0 = qb * 64 + w * 16;
  const _Float16* xb = xh + (size_t)b * SEQ * FDIM + h * KD;

  // Q fragment (A layout), raw f16 (log2e lives in the leaky constants)
  v16h aq;
  {
    const _Float16* qrow = xb + (size_t)(i0 + ln) * FDIM;
#pragma unroll
    for (int e = 0; e < 16; ++e)
      aq[e] = qrow[amapK(hi, e)];
  }

  v8f o_lo = {}, o_hi = {};
  float mrow[8], lsum[8];
#pragma unroll
  for (int r = 0; r < 8; ++r) { mrow[r] = NEGBIG; lsum[r] = 0.0f; }

  const unsigned lds0 = (unsigned)(size_t)&sx[0][16 * w][0];
  const unsigned lds1 = (unsigned)(size_t)&sx[1][16 * w][0];

  // prologue: each wave DMAs its quarter of chunk 0
  tdm_load_16x32(lds0, xb + (size_t)(16 * w) * FDIM);

  for (int jc = 0; jc < NCH; ++jc) {
    const int j0  = jc * JT;
    const int cur = jc & 1;
    const _Float16 (*sxc)[SXS] = sx[cur];

    __builtin_amdgcn_s_wait_tensorcnt(0);  // my quarter landed in LDS
    __syncthreads();                       // everyone's quarter landed

    // overlap: DMA next chunk into the other buffer while we compute
    if (jc + 1 < NCH)
      tdm_load_16x32(cur ? lds0 : lds1,
                     xb + (size_t)((jc + 1) * JT + 16 * w) * FDIM);

    // ---- QK^T: four 16x16 score tiles ----
    v8f c[4];
#pragma unroll
    for (int t = 0; t < 4; ++t) {
      v16h bk;
#pragma unroll
      for (int e = 0; e < 16; ++e)
        bk[e] = sxc[16 * t + ln][bmapK(hi, e)];
      v8f z = {};
      c[t] = __builtin_amdgcn_wmma_f32_16x16x32_f16(false, aq, false, bk,
                                                    (short)0, z, false, false);
    }

    // hoisted V-fragment DS loads (overlap with softmax VALU)
    v16h bvl0, bvl1, bvh0, bvh1;
#pragma unroll
    for (int e = 0; e < 16; ++e) {
      const int k = bmapK(hi, e);
      bvl0[e] = sxc[k][ln];
      bvh0[e] = sxc[k][ln + 16];
      bvl1[e] = sxc[32 + k][ln];
      bvh1[e] = sxc[32 + k][ln + 16];
    }

    // ---- leaky(+log2e) + mask + online softmax ----
#pragma unroll
    for (int r = 0; r < 8; ++r) {
      const int m = r + 8 * hi;
      const int* arow = adj + (size_t)(i0 + m) * SEQ + j0 + ln;
      const int a0 = arow[0];
      const int a1 = arow[16];
      const int a2 = arow[32];
      const int a3 = arow[48];

      float s0 = c[0][r]; s0 = fmaf(C2L, __builtin_fabsf(s0), C1L * s0);
      float s1 = c[1][r]; s1 = fmaf(C2L, __builtin_fabsf(s1), C1L * s1);
      float s2 = c[2][r]; s2 = fmaf(C2L, __builtin_fabsf(s2), C1L * s2);
      float s3 = c[3][r]; s3 = fmaf(C2L, __builtin_fabsf(s3), C1L * s3);
      float t0 = a0 ? s0 : NEGBIG;
      float t1 = a1 ? s1 : NEGBIG;
      float t2 = a2 ? s2 : NEGBIG;
      float t3 = a3 ? s3 : NEGBIG;

      float mc   = rowmax16(fmaxf(fmaxf(t0, t1), fmaxf(t2, t3)));
      float mnew = fmaxf(mrow[r], mc);
      float sc   = exp2f(mrow[r] - mnew);
      mrow[r] = mnew;

      float p0 = exp2f(t0 - mnew);
      float p1 = exp2f(t1 - mnew);
      float p2 = exp2f(t2 - mnew);
      float p3 = exp2f(t3 - mnew);
      lsum[r] = fmaf(lsum[r], sc, (p0 + p1) + (p2 + p3));
      o_lo[r] *= sc;
      o_hi[r] *= sc;

      sp[w][m][ln]      = (_Float16)p0;
      sp[w][m][ln + 16] = (_Float16)p1;
      sp[w][m][ln + 32] = (_Float16)p2;
      sp[w][m][ln + 48] = (_Float16)p3;
    }

    // ---- P·V ----
    v16h ap0, ap1;
#pragma unroll
    for (int e = 0; e < 16; ++e) {
      const int ka = amapK(hi, e);
      ap0[e] = sp[w][ln][ka];
      ap1[e] = sp[w][ln][32 + ka];
    }
    o_lo = __builtin_amdgcn_wmma_f32_16x16x32_f16(false, ap0, false, bvl0,
                                                  (short)0, o_lo, false, false);
    o_lo = __builtin_amdgcn_wmma_f32_16x16x32_f16(false, ap1, false, bvl1,
                                                  (short)0, o_lo, false, false);
    o_hi = __builtin_amdgcn_wmma_f32_16x16x32_f16(false, ap0, false, bvh0,
                                                  (short)0, o_hi, false, false);
    o_hi = __builtin_amdgcn_wmma_f32_16x16x32_f16(false, ap1, false, bvh1,
                                                  (short)0, o_hi, false, false);
    // no trailing barrier: next iteration's wait+barrier protects buffers
  }

  float* ob = out + (size_t)b * SEQ * FDIM + h * KD;
#pragma unroll
  for (int r = 0; r < 8; ++r) {
    const int m = r + 8 * hi;
    const float inv = __builtin_amdgcn_rcpf(rowsum16(lsum[r]));
    ob[(size_t)(i0 + m) * FDIM + ln]      = o_lo[r] * inv;
    ob[(size_t)(i0 + m) * FDIM + ln + 16] = o_hi[r] * inv;
  }
}

// ------------------- f32 -> f16 conversion pre-kernel ---------------------
__global__ __launch_bounds__(256)
void cvt_f32_f16_kernel(const float* __restrict__ in,
                        _Float16* __restrict__ o, int n4) {
  int i = blockIdx.x * blockDim.x + threadIdx.x;
  if (i < n4) {
    v4f f = ((const v4f*)in)[i];
    v4h hh;
#pragma unroll
    for (int q = 0; q < 4; ++q) hh[q] = (_Float16)f[q];
    ((v4h*)o)[i] = hh;
  }
}

// ------------------- fallback kernel (no workspace): round-3 path ---------
__global__ __launch_bounds__(128)
void gat_flash_wmma_gfx1250(const float* __restrict__ x,
                            const int*   __restrict__ adj,
                            float*       __restrict__ out) {
  __shared__ _Float16 sx[JT][SXS];
  __shared__ _Float16 sp[4][16][SPS];

  const int lane = threadIdx.x & 31;
  const int w    = threadIdx.x >> 5;
  const int hi   = lane >> 4;
  const int ln   = lane & 15;

  const int qb = blockIdx.x & 63;
  const int h  = (blockIdx.x >> 6) & (NH - 1);
  const int b  = blockIdx.x >> 8;

  const int i0 = qb * 64 + w * 16;
  const float* xb = x + (size_t)b * SEQ * FDIM + h * KD;

  v16h aq;
  {
    const float* qrow = xb + (size_t)(i0 + ln) * FDIM;
#pragma unroll
    for (int e = 0; e < 16; ++e)
      aq[e] = (_Float16)(qrow[amapK(hi, e)] * LOG2E);
  }

  v8f o_lo = {}, o_hi = {};
  float mrow[8], lsum[8];
#pragma unroll
  for (int r = 0; r < 8; ++r) { mrow[r] = NEGBIG; lsum[r] = 0.0f; }

  for (int jc = 0; jc < NCH; ++jc) {
    const int j0 = jc * JT;
    {
      const int t  = threadIdx.x;
      const int jr = t >> 1;
      const int dh = (t & 1) * 16;
      const float* src = xb + (size_t)(j0 + jr) * FDIM + dh;
      v4f f0 = *(const v4f*)(src + 0);
      v4f f1 = *(const v4f*)(src + 4);
      v4f f2 = *(const v4f*)(src + 8);
      v4f f3 = *(const v4f*)(src + 12);
      if (jc + 1 < NCH)
        __builtin_prefetch(src + (size_t)JT * FDIM, 0, 3);
      v4h h0, h1, h2, h3;
#pragma unroll
      for (int q = 0; q < 4; ++q) {
        h0[q] = (_Float16)f0[q];
        h1[q] = (_Float16)f1[q];
        h2[q] = (_Float16)f2[q];
        h3[q] = (_Float16)f3[q];
      }
      *(v4h*)&sx[jr][dh + 0]  = h0;
      *(v4h*)&sx[jr][dh + 4]  = h1;
      *(v4h*)&sx[jr][dh + 8]  = h2;
      *(v4h*)&sx[jr][dh + 12] = h3;
    }
    __syncthreads();

    v8f c[4];
#pragma unroll
    for (int t = 0; t < 4; ++t) {
      v16h bk;
#pragma unroll
      for (int e = 0; e < 16; ++e)
        bk[e] = sx[16 * t + ln][bmapK(hi, e)];
      v8f z = {};
      c[t] = __builtin_amdgcn_wmma_f32_16x16x32_f16(false, aq, false, bk,
                                                    (short)0, z, false, false);
    }

    v16h bvl0, bvl1, bvh0, bvh1;
#pragma unroll
    for (int e = 0; e < 16; ++e) {
      const int k = bmapK(hi, e);
      bvl0[e] = sx[k][ln];
      bvh0[e] = sx[k][ln + 16];
      bvl1[e] = sx[32 + k][ln];
      bvh1[e] = sx[32 + k][ln + 16];
    }

#pragma unroll
    for (int r = 0; r < 8; ++r) {
      const int m = r + 8 * hi;
      const int* arow = adj + (size_t)(i0 + m) * SEQ + j0 + ln;
      const int a0 = arow[0];
      const int a1 = arow[16];
      const int a2 = arow[32];
      const int a3 = arow[48];

      float s0 = c[0][r]; s0 = fmaf(C2, __builtin_fabsf(s0), C1 * s0);
      float s1 = c[1][r]; s1 = fmaf(C2, __builtin_fabsf(s1), C1 * s1);
      float s2 = c[2][r]; s2 = fmaf(C2, __builtin_fabsf(s2), C1 * s2);
      float s3 = c[3][r]; s3 = fmaf(C2, __builtin_fabsf(s3), C1 * s3);
      float t0 = a0 ? s0 : NEGBIG;
      float t1 = a1 ? s1 : NEGBIG;
      float t2 = a2 ? s2 : NEGBIG;
      float t3 = a3 ? s3 : NEGBIG;

      float mc   = rowmax16(fmaxf(fmaxf(t0, t1), fmaxf(t2, t3)));
      float mnew = fmaxf(mrow[r], mc);
      float sc   = exp2f(mrow[r] - mnew);
      mrow[r] = mnew;

      float p0 = exp2f(t0 - mnew);
      float p1 = exp2f(t1 - mnew);
      float p2 = exp2f(t2 - mnew);
      float p3 = exp2f(t3 - mnew);
      lsum[r] = fmaf(lsum[r], sc, (p0 + p1) + (p2 + p3));
      o_lo[r] *= sc;
      o_hi[r] *= sc;

      sp[w][m][ln]      = (_Float16)p0;
      sp[w][m][ln + 16] = (_Float16)p1;
      sp[w][m][ln + 32] = (_Float16)p2;
      sp[w][m][ln + 48] = (_Float16)p3;
    }

    v16h ap0, ap1;
#pragma unroll
    for (int e = 0; e < 16; ++e) {
      const int ka = amapK(hi, e);
      ap0[e] = sp[w][ln][ka];
      ap1[e] = sp[w][ln][32 + ka];
    }
    o_lo = __builtin_amdgcn_wmma_f32_16x16x32_f16(false, ap0, false, bvl0,
                                                  (short)0, o_lo, false, false);
    o_lo = __builtin_amdgcn_wmma_f32_16x16x32_f16(false, ap1, false, bvl1,
                                                  (short)0, o_lo, false, false);
    o_hi = __builtin_amdgcn_wmma_f32_16x16x32_f16(false, ap0, false, bvh0,
                                                  (short)0, o_hi, false, false);
    o_hi = __builtin_amdgcn_wmma_f32_16x16x32_f16(false, ap1, false, bvh1,
                                                  (short)0, o_hi, false, false);
    __syncthreads();
  }

  float* ob = out + (size_t)b * SEQ * FDIM + h * KD;
#pragma unroll
  for (int r = 0; r < 8; ++r) {
    const int m = r + 8 * hi;
    const float inv = __builtin_amdgcn_rcpf(rowsum16(lsum[r]));
    ob[(size_t)(i0 + m) * FDIM + ln]      = o_lo[r] * inv;
    ob[(size_t)(i0 + m) * FDIM + ln + 16] = o_hi[r] * inv;
  }
}

extern "C" void kernel_launch(void* const* d_in, const int* in_sizes, int n_in,
                              void* d_out, int out_size, void* d_ws, size_t ws_size,
                              hipStream_t stream) {
  (void)in_sizes; (void)n_in; (void)out_size;
  const float* x   = (const float*)d_in[0];   // (2, 4096, 128) f32
  const int*   adj = (const int*)d_in[1];     // (4096, 4096) i32
  float*       out = (float*)d_out;           // (2, 4096, 128) f32

  dim3 grid(2 * NH * (SEQ / 64));
  dim3 block(128);
  const size_t need = (size_t)2 * SEQ * FDIM * sizeof(_Float16);  // 2 MB
  if (d_ws != nullptr && ws_size >= need) {
    _Float16* xh = (_Float16*)d_ws;
    const int n4 = 2 * SEQ * FDIM / 4;
    hipLaunchKernelGGL(cvt_f32_f16_kernel, dim3((n4 + 255) / 256), dim3(256),
                       0, stream, x, xh, n4);
    hipLaunchKernelGGL(gat_flash_wmma_tdm, grid, block, 0, stream, xh, adj, out);
  } else {
    hipLaunchKernelGGL(gat_flash_wmma_gfx1250, grid, block, 0, stream,
                       x, adj, out);
  }
}
